// GINConv_88742614270550
// MI455X (gfx1250) — matrix operand order repaired and checked by reading, not compiled, not run on hardware
//
#include <hip/hip_runtime.h>

// ---------------------------------------------------------------------------
// GINConv edge MLP for gfx1250 (MI455X):
//   s = x[src] + x[dst]                [E,128] (gather; x = 51MB -> L2-resident)
//   h = relu(s @ W1^T + b1)            [E,256]
//   out = 0.5 * (h @ W2^T + b2)        [E,128]
// GEMMs on V_WMMA_F32_16X16X32_F16 (f16 in, f32 accumulate).
// Wave owns an N-slice and holds its B fragments in registers (4x reuse
// across the block's 4 M-strips); A fragments streamed from padded LDS.
// ---------------------------------------------------------------------------

typedef __attribute__((ext_vector_type(16))) _Float16 v16h;
typedef __attribute__((ext_vector_type(8)))  _Float16 v8h;
typedef __attribute__((ext_vector_type(8)))  float    v8f;
typedef __attribute__((ext_vector_type(4)))  float    v4f;

#define D_IN   128
#define D_HID  256
#define D_OUT  128
#define TILE_E 64            // edges per block (4 M-strips of 16)
#define BLOCK  128           // 4 waves of 32
#define SROW   (D_IN + 8)    // 136 halves/row: 272B = 68 dwords -> 4-bank rotate
#define HROW   (D_HID + 8)   // 264 halves/row: 528B = 132 dwords -> 4-bank rotate

__device__ __forceinline__ v16h join16(v8h lo, v8h hi) {
    union { v16h v; v8h h[2]; } u;
    u.h[0] = lo; u.h[1] = hi;
    return u.v;
}

// f32 -> f16 weight conversion (W1: 256x128, W2: 128x256; 64KB each in f16)
__global__ void cvt_w_f16(const float* __restrict__ src,
                          _Float16* __restrict__ dst, int n) {
    int i = blockIdx.x * blockDim.x + threadIdx.x;
    if (i < n) dst[i] = (_Float16)src[i];
}

__global__ __launch_bounds__(BLOCK)
void gin_edge_mlp(const float* __restrict__ x,
                  const long long* __restrict__ eidx,   // [2][E] int64
                  const _Float16* __restrict__ W1h,     // [256][128] row-major
                  const float* __restrict__ b1,         // [256]
                  const _Float16* __restrict__ W2h,     // [128][256] row-major
                  const float* __restrict__ b2,         // [128]
                  float* __restrict__ out,              // [E][128]
                  int E) {
    __shared__ _Float16 sA[TILE_E * SROW];   // 17408 B: s tile, f16
    __shared__ _Float16 sH[TILE_E * HROW];   // 33792 B: h tile, f16

    const int tid = threadIdx.x;
    const int e0  = blockIdx.x * TILE_E;

    // ---------------- Stage 1: gather + add + cvt -> sA ----------------
    {
        const int e    = tid >> 1;          // 0..63
        const int half = tid & 1;           // 64-feature half
        int ge = e0 + e;
        if (ge >= E) ge = E - 1;            // clamp (E % 64 == 0 anyway)
        const long long si = eidx[ge];
        const long long di = eidx[(size_t)E + ge];
        const float* __restrict__ xs = x + (size_t)si * D_IN + half * 64;
        const float* __restrict__ xd = x + (size_t)di * D_IN + half * 64;
        _Float16* dst = &sA[e * SROW + half * 64];
#pragma unroll
        for (int f = 0; f < 64; f += 8) {
            v4f a0 = *(const v4f*)(xs + f);
            v4f a1 = *(const v4f*)(xs + f + 4);
            v4f c0 = *(const v4f*)(xd + f);
            v4f c1 = *(const v4f*)(xd + f + 4);
            v8h p;
            p[0] = (_Float16)(a0[0] + c0[0]);
            p[1] = (_Float16)(a0[1] + c0[1]);
            p[2] = (_Float16)(a0[2] + c0[2]);
            p[3] = (_Float16)(a0[3] + c0[3]);
            p[4] = (_Float16)(a1[0] + c1[0]);
            p[5] = (_Float16)(a1[1] + c1[1]);
            p[6] = (_Float16)(a1[2] + c1[2]);
            p[7] = (_Float16)(a1[3] + c1[3]);
            *(v8h*)(dst + f) = p;
        }
    }
    __syncthreads();

    const int lane = tid & 31;
    const int wave = tid >> 5;        // 0..3
    const int l16  = lane & 15;       // column selector (N) / row-in-half (M)
    const int lh   = lane >> 4;       // half-wave id

    // ---------------- Stage 2: h = relu(s @ W1^T + b1) ----------------
    // Wave owns N-tiles [wave*4 .. wave*4+3] (64 hidden columns).
    // B fragments (16-bit B 32x16 layout): lane loads 16 contiguous halves of
    // row n0+l16 at K = kk*32 + lh*16. Held in registers; 4x reuse over M.
    v16h b1f[4][4];
    float bias1[4];
#pragma unroll
    for (int j = 0; j < 4; ++j) {
        const int n0 = (wave * 4 + j) * 16;
        bias1[j] = b1[n0 + l16];
#pragma unroll
        for (int kk = 0; kk < 4; ++kk)
            b1f[j][kk] = *(const v16h*)(W1h + (size_t)(n0 + l16) * D_IN
                                        + kk * 32 + lh * 16);
    }

#pragma unroll
    for (int m = 0; m < 4; ++m) {
        // A fragments (16-bit A 16x32 layout): lane holds row M = m*16+l16,
        // K = k0+lh*16..+15 (contiguous, matching B fragments).
        v16h af[4];
#pragma unroll
        for (int kk = 0; kk < 4; ++kk) {
            const _Float16* p = &sA[(m * 16 + l16) * SROW + kk * 32 + lh * 16];
            af[kk] = join16(*(const v8h*)p, *(const v8h*)(p + 8));
        }
#pragma unroll
        for (int j = 0; j < 4; ++j) {
            const float bv = bias1[j];
            v8f acc = {bv, bv, bv, bv, bv, bv, bv, bv};
#pragma unroll
            for (int kk = 0; kk < 4; ++kk)
                acc = __builtin_amdgcn_wmma_f32_16x16x32_f16(
                    false, af[kk], false, b1f[j][kk], (short)0, acc, false, false);
            // ReLU, cvt, scatter into sH (C layout: VGPR v -> m-row v + 8*lh)
            const int n0 = (wave * 4 + j) * 16;
#pragma unroll
            for (int v = 0; v < 8; ++v) {
                float r = acc[v] > 0.0f ? acc[v] : 0.0f;
                sH[(m * 16 + v + 8 * lh) * HROW + n0 + l16] = (_Float16)r;
            }
        }
    }
    __syncthreads();

    // ---------------- Stage 3: out = 0.5*(h @ W2^T + b2) ----------------
    // Wave owns N-tiles [wave*2, wave*2+1] (32 output columns).
    v16h b2f[2][8];
    float bias2[2];
#pragma unroll
    for (int j = 0; j < 2; ++j) {
        const int n0 = (wave * 2 + j) * 16;
        bias2[j] = b2[n0 + l16];
#pragma unroll
        for (int kk = 0; kk < 8; ++kk)
            b2f[j][kk] = *(const v16h*)(W2h + (size_t)(n0 + l16) * D_HID
                                        + kk * 32 + lh * 16);
    }

#pragma unroll
    for (int m = 0; m < 4; ++m) {
        v16h af[8];
#pragma unroll
        for (int kk = 0; kk < 8; ++kk) {
            const _Float16* p = &sH[(m * 16 + l16) * HROW + kk * 32 + lh * 16];
            af[kk] = join16(*(const v8h*)p, *(const v8h*)(p + 8));
        }
#pragma unroll
        for (int j = 0; j < 2; ++j) {
            const float bv = bias2[j];
            v8f acc = {bv, bv, bv, bv, bv, bv, bv, bv};
#pragma unroll
            for (int kk = 0; kk < 8; ++kk)
                acc = __builtin_amdgcn_wmma_f32_16x16x32_f16(
                    false, af[kk], false, b2f[j][kk], (short)0, acc, false, false);
            const int n0 = (wave * 2 + j) * 16;
#pragma unroll
            for (int v = 0; v < 8; ++v) {
                const int ge = e0 + m * 16 + v + 8 * lh;
                if (ge < E)
                    out[(size_t)ge * D_OUT + n0 + l16] = 0.5f * acc[v];
            }
        }
    }
}

extern "C" void kernel_launch(void* const* d_in, const int* in_sizes, int n_in,
                              void* d_out, int out_size, void* d_ws, size_t ws_size,
                              hipStream_t stream) {
    const float*     x    = (const float*)d_in[0];
    const long long* eidx = (const long long*)d_in[1];
    const float*     W1   = (const float*)d_in[2];
    const float*     b1   = (const float*)d_in[3];
    const float*     W2   = (const float*)d_in[4];
    const float*     b2   = (const float*)d_in[5];
    float*           out  = (float*)d_out;

    const int E = in_sizes[1] / 2;          // edge_index is [2, E]

    // scratch: f16 weights (64KB + 64KB)
    _Float16* W1h = (_Float16*)d_ws;
    _Float16* W2h = (_Float16*)((char*)d_ws + (size_t)D_HID * D_IN * sizeof(_Float16));

    const int nW = D_HID * D_IN;            // 32768 (same count both weights)
    cvt_w_f16<<<(nW + 255) / 256, 256, 0, stream>>>(W1, W1h, nW);
    cvt_w_f16<<<(nW + 255) / 256, 256, 0, stream>>>(W2, W2h, nW);

    const int nblocks = (E + TILE_E - 1) / TILE_E;   // 9375 for E=600000
    gin_edge_mlp<<<nblocks, BLOCK, 0, stream>>>(x, eidx, W1h, b1, W2h, b2, out, E);
}
